// SMARTAgentDecoder_43705587204551
// MI455X (gfx1250) — matrix operand order repaired and checked by reading, not compiled, not run on hardware
//
// SMART agent-decoder forward for MI455X (gfx1250).
// All dense math routed through V_WMMA_F32_16X16X32_BF16 (bf16 in, fp32 accum);
// LayerNorm / segment-softmax / geometry kept in fp32 VALU.
// GEMM: 128x64 block tile, 8 waves (2Mx4N), 4 WMMA accumulators per wave.
// Fast path (K%32==0): unclamped float2 A loads + global prefetch of next slab.
// Epilogue templated on (EPI, ACC, RES) with unguarded full-tile stores.
// Workspace requirement: ~460 MB (float scratch carved below).

#include <hip/hip_runtime.h>
#include <math.h>

// ---------------- model constants ----------------
#define MD   128          // feature dim D
#define MH   8            // heads
#define MDH  16           // head dim
#define MN   512          // agents N
#define MS   18           // steps S
#define MNT  (MN*MS)      // 9216 tokens
#define MNPL 8192
#define MTOK 512
#define MF   64           // fourier freqs
#define PI_F 3.14159265358979323846f

typedef __attribute__((ext_vector_type(16))) __bf16 v16bf;
typedef __attribute__((ext_vector_type(8)))  float  v8f;

// ---------------- small device helpers ----------------
__device__ __forceinline__ unsigned pack_bf16(float a, float b) {
  union { float f; unsigned u; } ua, ub; ua.f = a; ub.f = b;
  unsigned ha = (ua.u + 0x7FFFu + ((ua.u >> 16) & 1u)) >> 16;
  unsigned hb = (ub.u + 0x7FFFu + ((ub.u >> 16) & 1u)) >> 16;
  return (ha & 0xFFFFu) | (hb << 16);
}
// monotonic float<->uint encoding for atomic max on floats
__device__ __forceinline__ unsigned fenc(float f) {
  unsigned u = __float_as_uint(f);
  return (u & 0x80000000u) ? ~u : (u | 0x80000000u);
}
__device__ __forceinline__ float fdec(unsigned u) {
  return (u & 0x80000000u) ? __uint_as_float(u & 0x7FFFFFFFu) : __uint_as_float(~u);
}
__device__ __forceinline__ float wrapf(float a) {
  float t = fmodf(a + PI_F, 2.0f * PI_F);
  if (t < 0.f) t += 2.0f * PI_F;
  return t - PI_F;
}
__device__ __forceinline__ float block_reduce128(float v, float* red) {
  int t = threadIdx.x;
  red[t] = v; __syncthreads();
  for (int s = 64; s > 0; s >>= 1) { if (t < s) red[t] += red[t + s]; __syncthreads(); }
  float r = red[0]; __syncthreads();
  return r;
}

// ---------------- WMMA fragment loads (contiguous k = 16h+s order, same for A and B) --------
// A 16x32 bf16: lane-half g: K [16g..16g+16) in u32s [8g..8g+8) (matches fragB)
__device__ __forceinline__ v16bf fragA(const unsigned (*Sm)[17], int base, int lane) {
  int r = base + (lane & 15);
  int g = (lane >> 4) * 8;
  union { v16bf v; unsigned u[8]; } t;
#pragma unroll
  for (int j = 0; j < 4; ++j) t.u[j] = Sm[r][g + j];
#pragma unroll
  for (int j = 0; j < 4; ++j) t.u[4 + j] = Sm[r][4 + g + j];
  return t.v;
}
// B 32x16 bf16 (stored n-major transposed in LDS): lanes 0-15 K=0..15, 16-31 K=16..31
__device__ __forceinline__ v16bf fragB(const unsigned (*Sm)[17], int base, int lane) {
  int r = base + (lane & 15);
  int g = (lane >> 4) * 8;
  union { v16bf v; unsigned u[8]; } t;
#pragma unroll
  for (int j = 0; j < 8; ++j) t.u[j] = Sm[r][g + j];
  return t.v;
}

// ---------------- generic WMMA GEMM ----------------
// C[M,N] = EPI( A[M,K] @ B[K,N] + bias + (ACC? C) + (RES? resid) )
// EPI: 0 none, 1 relu, 2 sigmoid.  Block tile 128x64, 8 waves (2Mx4N),
// each wave computes 64x16 via 4 WMMA accumulators.
template<int EPI, bool ACC, bool RES>
__global__ void __launch_bounds__(256)
sk_gemm(const float* __restrict__ A, const float* __restrict__ B,
        const float* __restrict__ bias, const float* __restrict__ resid,
        float* __restrict__ C, int M, int N, int K) {
  __shared__ unsigned As[128][17];
  __shared__ unsigned Bs[64][17];
  const int tid = threadIdx.x;
  const int tileM = blockIdx.y * 128, tileN = blockIdx.x * 64;
  const int lane = tid & 31, wid = tid >> 5;
  const int mw = (wid & 1) * 64, nw = (wid >> 1) * 16;
  const v8f vzero = {0.f,0.f,0.f,0.f,0.f,0.f,0.f,0.f};
  v8f acc[4] = {vzero, vzero, vzero, vzero};

  if ((K & 31) == 0) {
    // ---- fast path: K multiple of 32, no clamping, vector A loads ----
    for (int kk = 0; kk < K; kk += 32) {
#pragma unroll
      for (int i = 0; i < 8; ++i) {         // A slab: 128 rows x 32 k
        int li = tid + i * 256;
        int m = li >> 4, j = li & 15;
        int row = tileM + m; row = (row < M) ? row : (M - 1);
        const float* ap = A + (size_t)row * K + kk + 2 * j;
        float2 a = *(const float2*)ap;      // 8B-aligned: K, kk, 2j all even
        As[m][j] = pack_bf16(a.x, a.y);
        if (kk + 32 < K) __builtin_prefetch(ap + 32, 0, 3);
      }
#pragma unroll
      for (int i = 0; i < 4; ++i) {         // B slab transposed: 64 cols x 32 k
        int li = tid + i * 256;
        int j = li >> 6, n = li & 63;
        int col = tileN + n; col = (col < N) ? col : (N - 1);
        const float* bp = B + (size_t)(kk + 2 * j) * N + col;
        Bs[n][j] = pack_bf16(bp[0], bp[N]);
        if (kk + 32 < K) __builtin_prefetch(bp + (size_t)32 * N, 0, 3);
      }
      __syncthreads();
      v16bf bb = fragB(Bs, nw, lane);
#pragma unroll
      for (int t = 0; t < 4; ++t) {
        v16bf a = fragA(As, mw + t * 16, lane);
        acc[t] = __builtin_amdgcn_wmma_f32_16x16x32_bf16(false, a, false, bb,
                                                         (short)0, acc[t], false, false);
      }
      __syncthreads();
    }
  } else {
    // ---- generic path (K=129): clamped indices, select-zeroed tails ----
    for (int kk = 0; kk < K; kk += 32) {
#pragma unroll
      for (int i = 0; i < 8; ++i) {
        int li = tid + i * 256;
        int m = li >> 4, j = li & 15;
        int row = tileM + m; row = (row < M) ? row : (M - 1);
        int k0 = kk + 2 * j;
        int kc0 = (k0 < K) ? k0 : 0;
        int kc1 = (k0 + 1 < K) ? (k0 + 1) : 0;
        float a0 = A[(size_t)row * K + kc0];
        float a1 = A[(size_t)row * K + kc1];
        a0 = (k0     < K) ? a0 : 0.f;
        a1 = (k0 + 1 < K) ? a1 : 0.f;
        As[m][j] = pack_bf16(a0, a1);
      }
#pragma unroll
      for (int i = 0; i < 4; ++i) {
        int li = tid + i * 256;
        int j = li >> 6, n = li & 63;
        int col = tileN + n; col = (col < N) ? col : (N - 1);
        int k0 = kk + 2 * j;
        int kc0 = (k0 < K) ? k0 : 0;
        int kc1 = (k0 + 1 < K) ? (k0 + 1) : 0;
        float b0 = B[(size_t)kc0 * N + col];
        float b1 = B[(size_t)kc1 * N + col];
        b0 = (k0     < K) ? b0 : 0.f;
        b1 = (k0 + 1 < K) ? b1 : 0.f;
        Bs[n][j] = pack_bf16(b0, b1);
      }
      __syncthreads();
      v16bf bb = fragB(Bs, nw, lane);
#pragma unroll
      for (int t = 0; t < 4; ++t) {
        v16bf a = fragA(As, mw + t * 16, lane);
        acc[t] = __builtin_amdgcn_wmma_f32_16x16x32_bf16(false, a, false, bb,
                                                         (short)0, acc[t], false, false);
      }
      __syncthreads();
    }
  }

  // ---- epilogue ----
  const int n = tileN + nw + (lane & 15);
  const bool nok = (n < N);
  const float bv = (bias && nok) ? bias[n] : 0.f;
  if (nok && (tileM + 128 <= M)) {
    // full tile: unguarded stores
#pragma unroll
    for (int t = 0; t < 4; ++t) {
      const int mb = tileM + mw + t * 16 + (lane >> 4) * 8;
#pragma unroll
      for (int r = 0; r < 8; ++r) {
        size_t o = (size_t)(mb + r) * N + n;
        float x = acc[t][r] + bv;
        if (ACC) x += C[o];
        if (RES) x += resid[o];
        if (EPI == 1) x = fmaxf(x, 0.f);
        if (EPI == 2) x = 1.f / (1.f + __expf(-x));
        C[o] = x;
      }
    }
  } else if (nok) {
#pragma unroll
    for (int t = 0; t < 4; ++t) {
      const int mb = tileM + mw + t * 16 + (lane >> 4) * 8;
#pragma unroll
      for (int r = 0; r < 8; ++r) {
        int m0 = mb + r;
        if (m0 < M) {
          size_t o = (size_t)m0 * N + n;
          float x = acc[t][r] + bv;
          if (ACC) x += C[o];
          if (RES) x += resid[o];
          if (EPI == 1) x = fmaxf(x, 0.f);
          if (EPI == 2) x = 1.f / (1.f + __expf(-x));
          C[o] = x;
        }
      }
    }
  }
}

// ---------------- row LayerNorm over 128 dims ----------------
__global__ void sk_ln(const float* __restrict__ x, const float* __restrict__ w,
                      const float* __restrict__ b, float* __restrict__ out,
                      int nrows, int do_relu) {
  __shared__ float red[128];
  int row = blockIdx.x;
  if (row >= nrows) return;
  int t = threadIdx.x;
  float v = x[(size_t)row * MD + t];
  float mu = block_reduce128(v, red) * (1.f / MD);
  float d = v - mu;
  float var = block_reduce128(d * d, red) * (1.f / MD);
  float y = d * rsqrtf(var + 1e-5f) * w[t] + b[t];
  if (do_relu) y = fmaxf(y, 0.f);
  out[(size_t)row * MD + t] = y;
}

// ---------------- fills / elementwise ----------------
__global__ void sk_fill_f(float* p, size_t n, float v) {
  size_t i = (size_t)blockIdx.x * blockDim.x + threadIdx.x;
  if (i < n) p[i] = v;
}
__global__ void sk_fill_u(unsigned* p, size_t n, unsigned v) {
  size_t i = (size_t)blockIdx.x * blockDim.x + threadIdx.x;
  if (i < n) p[i] = v;
}
__global__ void sk_combine(const float* agg, const float* g, const float* s,
                           float* out, size_t n) {
  size_t i = (size_t)blockIdx.x * blockDim.x + threadIdx.x;
  if (i < n) { float a = agg[i]; out[i] = a + g[i] * (s[i] - a); }
}
__global__ void sk_concat2(const float* a, const float* b, float* out, int nrows) {
  size_t i = (size_t)blockIdx.x * blockDim.x + threadIdx.x;
  if (i >= (size_t)nrows * 256) return;
  size_t row = i >> 8; int d = (int)(i & 255);
  out[i] = (d < MD) ? a[row * MD + d] : b[row * MD + (d - MD)];
}
__global__ void sk_add_cat(float* acc, const float* cat, int nrows, int srep) {
  size_t i = (size_t)blockIdx.x * blockDim.x + threadIdx.x;
  if (i >= (size_t)nrows * MD) return;
  size_t row = i >> 7; int d = (int)(i & 127);
  acc[i] += cat[(row / srep) * MD + d];
}

// ---------------- fourier feature generation (one channel) ----------------
__global__ void sk_fourier_feat(const float* __restrict__ feats, int n, int C, int c,
                                const float* __restrict__ freqs, float* __restrict__ out) {
  size_t i = (size_t)blockIdx.x * blockDim.x + threadIdx.x;
  if (i >= (size_t)n * 129) return;
  int row = (int)(i / 129), j = (int)(i % 129);
  float x = feats[(size_t)row * C + c];
  float r;
  if (j < 2 * MF) {
    float f = x * freqs[c * MF + (j & (MF - 1))] * (2.0f * PI_F);
    r = (j < MF) ? cosf(f) : sinf(f);
  } else r = x;
  out[i] = r;
}

// ---------------- feature kernels ----------------
__global__ void sk_motion_feat(const float* __restrict__ pos, const float* __restrict__ head,
                               float* __restrict__ out) {
  int row = blockIdx.x * blockDim.x + threadIdx.x;
  if (row >= MNT) return;
  int s = row % MS;
  float mvx = 0.f, mvy = 0.f;
  if (s > 0) { mvx = pos[2 * row] - pos[2 * (row - 1)];
               mvy = pos[2 * row + 1] - pos[2 * (row - 1) + 1]; }
  float hd = head[row], cx = cosf(hd), cy = sinf(hd);
  out[row * 2 + 0] = sqrtf(mvx * mvx + mvy * mvy + 1e-12f);
  out[row * 2 + 1] = atan2f(cx * mvy - cy * mvx, cx * mvx + cy * mvy + 1e-6f);
}
__global__ void sk_edge_t_feat(const int* __restrict__ edges, int E,
                               const float* __restrict__ pos, const float* __restrict__ head,
                               float* __restrict__ out) {
  int e = blockIdx.x * blockDim.x + threadIdx.x;
  if (e >= E) return;
  int s = edges[e], d = edges[E + e];
  float rx = pos[2 * s] - pos[2 * d], ry = pos[2 * s + 1] - pos[2 * d + 1];
  float hd = head[d], cx = cosf(hd), cy = sinf(hd);
  out[e * 4 + 0] = sqrtf(rx * rx + ry * ry + 1e-12f);
  out[e * 4 + 1] = atan2f(cx * ry - cy * rx, cx * rx + cy * ry + 1e-6f);
  out[e * 4 + 2] = wrapf(head[s] - hd);
  out[e * 4 + 3] = (float)((d % MS) - (s % MS));
}
__global__ void sk_edge_p_feat(const int* __restrict__ edges, int E,
                               const float* __restrict__ pos_pl, const float* __restrict__ orient_pl,
                               const float* __restrict__ pos, const float* __restrict__ head,
                               float* __restrict__ out) {
  int e = blockIdx.x * blockDim.x + threadIdx.x;
  if (e >= E) return;
  int s = edges[e], d = edges[E + e];
  float rx = pos_pl[2 * s] - pos[2 * d], ry = pos_pl[2 * s + 1] - pos[2 * d + 1];
  float hd = head[d], cx = cosf(hd), cy = sinf(hd);
  out[e * 3 + 0] = sqrtf(rx * rx + ry * ry + 1e-12f);
  out[e * 3 + 1] = atan2f(cx * ry - cy * rx, cx * rx + cy * ry + 1e-6f);
  out[e * 3 + 2] = wrapf(orient_pl[s] - hd);
}
__global__ void sk_edge_a_feat(const int* __restrict__ edges, int E,
                               const float* __restrict__ pos, const float* __restrict__ head,
                               float* __restrict__ out) {
  int e = blockIdx.x * blockDim.x + threadIdx.x;
  if (e >= E) return;
  int s = edges[e], d = edges[E + e];
  float rx = pos[2 * s] - pos[2 * d], ry = pos[2 * s + 1] - pos[2 * d + 1];
  float hd = head[d], cx = cosf(hd), cy = sinf(hd);
  out[e * 3 + 0] = sqrtf(rx * rx + ry * ry + 1e-12f);
  out[e * 3 + 1] = atan2f(cx * ry - cy * rx, cx * rx + cy * ry + 1e-6f);
  out[e * 3 + 2] = wrapf(head[s] - hd);
}

// ---------------- token table (3*512 rows, 8->128->128 MLP) ----------------
__global__ void sk_tok_table(const float* tv, const float* tp, const float* tc,
                             const float* W1, const float* b1,
                             const float* lnw, const float* lnb,
                             const float* W2, const float* b2, float* tab) {
  __shared__ float feat[8];
  __shared__ float red[128];
  __shared__ float hsh[128];
  int row = blockIdx.x, t = threadIdx.x;
  int kt = row >> 9, tokid = row & 511;
  const float* src = (kt == 0) ? tv : (kt == 1) ? tp : tc;
  if (t < 8) feat[t] = src[tokid * 8 + t];
  __syncthreads();
  float h = b1[kt * MD + t];
#pragma unroll
  for (int f = 0; f < 8; ++f) h += feat[f] * W1[(size_t)kt * 8 * MD + f * MD + t];
  float mu = block_reduce128(h, red) * (1.f / MD);
  float d = h - mu;
  float var = block_reduce128(d * d, red) * (1.f / MD);
  float y = fmaxf(d * rsqrtf(var + 1e-5f) * lnw[kt * MD + t] + lnb[kt * MD + t], 0.f);
  hsh[t] = y; __syncthreads();
  float o = b2[kt * MD + t];
  for (int m = 0; m < MD; ++m) o += hsh[m] * W2[(size_t)kt * MD * MD + m * MD + t];
  tab[(size_t)row * MD + t] = o;
}

// ---------------- per-agent categorical sum: type_emb + shape MLP ----------
__global__ void sk_shape_cat(const float* shape, const int* atype, const float* type_emb,
                             const float* W1, const float* b1,
                             const float* lnw, const float* lnb,
                             const float* W2, const float* b2, float* out) {
  __shared__ float feat[3];
  __shared__ float red[128];
  __shared__ float hsh[128];
  int a = blockIdx.x, t = threadIdx.x;
  if (t < 3) feat[t] = shape[a * 3 + t];
  __syncthreads();
  float h = b1[t];
#pragma unroll
  for (int f = 0; f < 3; ++f) h += feat[f] * W1[f * MD + t];
  float mu = block_reduce128(h, red) * (1.f / MD);
  float d = h - mu;
  float var = block_reduce128(d * d, red) * (1.f / MD);
  float y = fmaxf(d * rsqrtf(var + 1e-5f) * lnw[t] + lnb[t], 0.f);
  hsh[t] = y; __syncthreads();
  float o = b2[t];
  for (int m = 0; m < MD; ++m) o += hsh[m] * W2[m * MD + t];
  out[(size_t)a * MD + t] = type_emb[atype[a] * MD + t] + o;
}

// ---------------- fusion input: concat(token_emb_gather, x_a) --------------
__global__ void sk_fuse_concat(const float* tab, const int* atype, const int* ati,
                               const float* xa, float* out) {
  size_t i = (size_t)blockIdx.x * blockDim.x + threadIdx.x;
  if (i >= (size_t)MNT * 256) return;
  size_t row = i >> 8; int d = (int)(i & 255);
  if (d < MD) {
    int agent = (int)(row / MS);
    int idx = atype[agent] * MTOK + ati[row];
    out[i] = tab[(size_t)idx * MD + d];
  } else out[i] = xa[row * MD + (d - MD)];
}

// ---------------- attention: scores / softmax / aggregate ------------------
__global__ void sk_sim(const float* __restrict__ q, const float* __restrict__ k,
                       const float* __restrict__ kr, const int* __restrict__ edges,
                       int E, float* __restrict__ sim, unsigned* __restrict__ menc) {
  int i = blockIdx.x * blockDim.x + threadIdx.x;
  if (i >= E * MH) return;
  int e = i >> 3, h = i & 7;
  int s = edges[e], d = edges[E + e];
  const float* qp = q + (size_t)d * MD + h * MDH;
  const float* kp = k + (size_t)s * MD + h * MDH;
  const float* rp = kr + (size_t)e * MD + h * MDH;
  float acc = 0.f;
#pragma unroll
  for (int j = 0; j < MDH; ++j) acc += qp[j] * (kp[j] + rp[j]);
  acc *= 0.25f;                                   // 1/sqrt(16)
  sim[i] = acc;
  atomicMax(&menc[(size_t)d * MH + h], fenc(acc));
}
__global__ void sk_expsum(float* __restrict__ sim, const unsigned* __restrict__ menc,
                          const int* __restrict__ edges, int E, float* __restrict__ den) {
  int i = blockIdx.x * blockDim.x + threadIdx.x;
  if (i >= E * MH) return;
  int e = i >> 3, h = i & 7;
  int d = edges[E + e];
  float ex = __expf(sim[i] - fdec(menc[(size_t)d * MH + h]));
  sim[i] = ex;
  atomicAdd(&den[(size_t)d * MH + h], ex);
}
__global__ void sk_aggregate(const float* __restrict__ sim, const float* __restrict__ den,
                             const float* __restrict__ v, const float* __restrict__ vr,
                             const int* __restrict__ edges, int E, float* __restrict__ agg) {
  size_t i = (size_t)blockIdx.x * blockDim.x + threadIdx.x;
  if (i >= (size_t)E * MD) return;
  int e = (int)(i >> 7), d = (int)(i & 127), h = d >> 4;
  int sN = edges[e], dN = edges[E + e];
  float w = sim[(size_t)e * MH + h] / (den[(size_t)dN * MH + h] + 1e-9f);
  atomicAdd(&agg[(size_t)dN * MD + d],
            (v[(size_t)sN * MD + d] + vr[(size_t)e * MD + d]) * w);
}

// =====================================================================
//                              host side
// =====================================================================
namespace {
struct AttnP { const float *Wff1,*Wff2,*Wg,*Wk,*Wkr,*Wo,*Wq,*Ws,*Wv,*Wvr,
                           *bff1,*bff2,*bg,*bo,*bq,*bs,
                           *ln_dst_b,*ln_dst_w,*ln_ff_b,*ln_ff_w,
                           *ln_r_b,*ln_r_w,*ln_src_b,*ln_src_w; };
struct FourP { const float *W1,*W2,*Wo,*b1,*b2,*bo,*freqs,*ln1b,*ln1w,*lnob,*lnow; };
struct MlpP  { const float *W1,*W2,*b1,*b2,*lnb,*lnw; };
}

extern "C" void kernel_launch(void* const* d_in, const int* in_sizes, int n_in,
                              void* d_out, int out_size, void* d_ws, size_t ws_size,
                              hipStream_t stream) {
  // ---------------- inputs (setup_inputs dict order) ----------------
  const float* pos_a     = (const float*)d_in[0];    // [N,S,2]
  const float* head_a    = (const float*)d_in[1];    // [N,S]
  const float* ashape    = (const float*)d_in[2];    // [N,3]
  const float* x_pl      = (const float*)d_in[3];    // [NPL,D]
  const float* pos_pl    = (const float*)d_in[4];    // [NPL,2]
  const float* orient_pl = (const float*)d_in[5];    // [NPL]
  const float* tok_veh   = (const float*)d_in[6];
  const float* tok_ped   = (const float*)d_in[7];
  const float* tok_cyc   = (const float*)d_in[8];
  const int*   atype     = (const int*)d_in[9];
  const int*   ati       = (const int*)d_in[10];
  const int*   e_t       = (const int*)d_in[11];
  const int*   e_p       = (const int*)d_in[12];
  const int*   e_a       = (const int*)d_in[13];
  const int ET = in_sizes[11] / 2, EP = in_sizes[12] / 2, EA = in_sizes[13] / 2;

  // ---------------- params: jax tree_flatten (sorted keys) ----------------
  int pi = 14;
  auto F = [&](int& i) { return (const float*)d_in[i++]; };
  auto getA = [&](int& i) { AttnP a;
    a.Wff1=F(i); a.Wff2=F(i); a.Wg=F(i); a.Wk=F(i); a.Wkr=F(i); a.Wo=F(i);
    a.Wq=F(i); a.Ws=F(i); a.Wv=F(i); a.Wvr=F(i);
    a.bff1=F(i); a.bff2=F(i); a.bg=F(i); a.bo=F(i); a.bq=F(i); a.bs=F(i);
    a.ln_dst_b=F(i); a.ln_dst_w=F(i); a.ln_ff_b=F(i); a.ln_ff_w=F(i);
    a.ln_r_b=F(i); a.ln_r_w=F(i); a.ln_src_b=F(i); a.ln_src_w=F(i); return a; };
  auto getF4 = [&](int& i) { FourP f;
    f.W1=F(i); f.W2=F(i); f.Wo=F(i); f.b1=F(i); f.b2=F(i); f.bo=F(i);
    f.freqs=F(i); f.ln1b=F(i); f.ln1w=F(i); f.lnob=F(i); f.lnow=F(i); return f; };
  auto getM = [&](int& i) { MlpP m;
    m.W1=F(i); m.W2=F(i); m.b1=F(i); m.b2=F(i); m.lnb=F(i); m.lnw=F(i); return m; };

  AttnP aaA   = getA(pi);
  MlpP  fusP  = getM(pi);
  MlpP  headP = getM(pi);
  AttnP ptA   = getA(pi);
  FourP raP   = getF4(pi);
  FourP rpP   = getF4(pi);
  FourP rtP   = getF4(pi);
  MlpP  shpP  = getM(pi);
  AttnP ttA   = getA(pi);
  MlpP  tokP  = getM(pi);
  const float* type_emb = F(pi);
  FourP xaP   = getF4(pi);

  // ---------------- workspace carving (floats) ----------------
  float* wsf = (float*)d_ws;
  size_t off = 0;
  auto alloc = [&](size_t n) { float* p = wsf + off; off += (n + 63) & ~(size_t)63; return p; };
  const int EMAXe = (EA > EP) ? ((EA > ET) ? EA : ET) : ((EP > ET) ? EP : ET);
  const int EM = (EMAXe > MNT) ? EMAXe : MNT;

  float* f_tab  = alloc((size_t)3 * MTOK * MD);
  float* f_cat  = alloc((size_t)MN * MD);
  float* f_mf   = alloc((size_t)MNT * 2);
  float* f_ef   = alloc((size_t)EM * 4);
  float* f_feat = alloc((size_t)EM * 129);   // fourier features; later reused as LN(r)
  float* f_h1   = alloc((size_t)EM * MD);    // fourier hidden;  later reused as kr
  float* f_h2   = alloc((size_t)EM * MD);    // fourier accum;   later reused as vr
  float* f_xa   = alloc((size_t)MNT * MD);
  float* f_x    = alloc((size_t)MNT * MD);
  float* f_rt   = alloc((size_t)ET * MD);
  float* f_rp   = alloc((size_t)EP * MD);
  float* f_ra   = alloc((size_t)EA * MD);
  float* f_xs   = alloc((size_t)MNT * MD);
  float* f_xd   = alloc((size_t)MNT * MD);
  float* f_q    = alloc((size_t)MNT * MD);
  float* f_k    = alloc((size_t)MNT * MD);
  float* f_v    = alloc((size_t)MNT * MD);
  float* f_sim  = alloc((size_t)EM * MH);
  float* f_den  = alloc((size_t)MNT * MH);
  unsigned* u_m = (unsigned*)alloc((size_t)MNT * MH);
  float* f_agg  = alloc((size_t)MNT * MD);
  float* f_g    = alloc((size_t)MNT * MD);
  float* f_s    = alloc((size_t)MNT * MD);
  float* f_o    = alloc((size_t)MNT * MD);
  float* f_xn   = alloc((size_t)MNT * MD);
  float* f_c2   = alloc((size_t)MNT * 256);
  float* f_ff   = alloc((size_t)MNT * 512);
  float* f_rn   = f_feat;                    // reuse after fourier phase
  float* f_kr   = f_h1;
  float* f_vr   = f_h2;
  (void)ws_size; (void)n_in; (void)out_size;

  // ---------------- launch helpers ----------------
  // epi: 0 none, 1 relu, 2 sigmoid; accum / resid dispatched to templates
  auto gemm = [&](const float* A, const float* B, const float* bias, const float* resid,
                  float* C, int M, int N, int K, int epi, int accum) {
    dim3 g((N + 63) / 64, (M + 127) / 128);
    if (epi == 1)
      sk_gemm<1,false,false><<<g, 256, 0, stream>>>(A, B, bias, resid, C, M, N, K);
    else if (epi == 2)
      sk_gemm<2,false,false><<<g, 256, 0, stream>>>(A, B, bias, resid, C, M, N, K);
    else if (accum)
      sk_gemm<0,true,false><<<g, 256, 0, stream>>>(A, B, bias, resid, C, M, N, K);
    else if (resid)
      sk_gemm<0,false,true><<<g, 256, 0, stream>>>(A, B, bias, resid, C, M, N, K);
    else
      sk_gemm<0,false,false><<<g, 256, 0, stream>>>(A, B, bias, resid, C, M, N, K);
  };
  auto ln = [&](const float* x, const float* w, const float* b, float* out, int n, int relu) {
    sk_ln<<<n, 128, 0, stream>>>(x, w, b, out, n, relu);
  };
  auto fillf = [&](float* p, size_t n, float v) {
    sk_fill_f<<<(unsigned)((n + 255) / 256), 256, 0, stream>>>(p, n, v);
  };
  auto fillu = [&](unsigned* p, size_t n, unsigned v) {
    sk_fill_u<<<(unsigned)((n + 255) / 256), 256, 0, stream>>>(p, n, v);
  };

  // FourierEmbedding: feats[n,C] -> out[n,128]
  auto fourier = [&](const float* feats, int n, int C, const FourP& P,
                     const float* cat, int srep, float* out) {
    for (int c = 0; c < C; ++c) {
      size_t tot = (size_t)n * 129;
      sk_fourier_feat<<<(unsigned)((tot + 255) / 256), 256, 0, stream>>>(feats, n, C, c, P.freqs, f_feat);
      gemm(f_feat, P.W1 + (size_t)c * 129 * MD, P.b1 + c * MD, nullptr, f_h1, n, MD, 129, 0, 0);
      ln(f_h1, P.ln1w + c * MD, P.ln1b + c * MD, f_h1, n, 1);
      gemm(f_h1, P.W2 + (size_t)c * MD * MD, P.b2 + c * MD, nullptr, f_h2, n, MD, MD, 0, c > 0);
    }
    if (cat) sk_add_cat<<<(unsigned)(((size_t)n * MD + 255) / 256), 256, 0, stream>>>(f_h2, cat, n, srep);
    ln(f_h2, P.lnow, P.lnob, f_h1, n, 1);
    gemm(f_h1, P.Wo, P.bo, nullptr, out, n, MD, MD, 0, 0);
  };

  // one AttentionLayer call; x (= f_x, NT rows) updated in place
  auto attn = [&](const float* xsrc, int nsrc, float* x, const float* r,
                  const int* edges, int E, const AttnP& P, int li) {
    const size_t o1 = (size_t)li * MD, oW = (size_t)li * MD * MD;
    ln(xsrc, P.ln_src_w + o1, P.ln_src_b + o1, f_xs, nsrc, 0);
    ln(x,    P.ln_dst_w + o1, P.ln_dst_b + o1, f_xd, MNT, 0);
    ln(r,    P.ln_r_w   + o1, P.ln_r_b   + o1, f_rn, E, 0);
    gemm(f_xd, P.Wq  + oW, P.bq + o1, nullptr, f_q,  MNT,  MD, MD, 0, 0);
    gemm(f_xs, P.Wk  + oW, nullptr,   nullptr, f_k,  nsrc, MD, MD, 0, 0);
    gemm(f_xs, P.Wv  + oW, nullptr,   nullptr, f_v,  nsrc, MD, MD, 0, 0);
    gemm(f_rn, P.Wkr + oW, nullptr,   nullptr, f_kr, E,    MD, MD, 0, 0);
    gemm(f_rn, P.Wvr + oW, nullptr,   nullptr, f_vr, E,    MD, MD, 0, 0);
    fillu(u_m, (size_t)MNT * MH, 0u);
    fillf(f_den, (size_t)MNT * MH, 0.f);
    fillf(f_agg, (size_t)MNT * MD, 0.f);
    sk_sim<<<(E * MH + 255) / 256, 256, 0, stream>>>(f_q, f_k, f_kr, edges, E, f_sim, u_m);
    sk_expsum<<<(E * MH + 255) / 256, 256, 0, stream>>>(f_sim, u_m, edges, E, f_den);
    sk_aggregate<<<(unsigned)(((size_t)E * MD + 255) / 256), 256, 0, stream>>>(
        f_sim, f_den, f_v, f_vr, edges, E, f_agg);
    sk_concat2<<<(unsigned)(((size_t)MNT * 256 + 255) / 256), 256, 0, stream>>>(f_agg, f_xd, f_c2, MNT);
    gemm(f_c2, P.Wg + (size_t)li * 256 * MD, P.bg + o1, nullptr, f_g, MNT, MD, 256, 2, 0);
    gemm(f_xd, P.Ws + oW, P.bs + o1, nullptr, f_s, MNT, MD, MD, 0, 0);
    sk_combine<<<(unsigned)(((size_t)MNT * MD + 255) / 256), 256, 0, stream>>>(
        f_agg, f_g, f_s, f_o, (size_t)MNT * MD);
    gemm(f_o, P.Wo + oW, P.bo + o1, x, f_xn, MNT, MD, MD, 0, 0);      // x + out@Wo+bo
    ln(f_xn, P.ln_ff_w + o1, P.ln_ff_b + o1, f_xd, MNT, 0);
    gemm(f_xd, P.Wff1 + (size_t)li * MD * 4 * MD, P.bff1 + (size_t)li * 4 * MD,
         nullptr, f_ff, MNT, 4 * MD, MD, 1, 0);
    gemm(f_ff, P.Wff2 + (size_t)li * 4 * MD * MD, P.bff2 + o1, f_xn, x, MNT, MD, 4 * MD, 0, 0);
  };

  // ---------------- pipeline ----------------
  // token tables + categorical sums
  sk_tok_table<<<3 * MTOK, 128, 0, stream>>>(tok_veh, tok_ped, tok_cyc,
      tokP.W1, tokP.b1, tokP.lnw, tokP.lnb, tokP.W2, tokP.b2, f_tab);
  sk_shape_cat<<<MN, 128, 0, stream>>>(ashape, atype, type_emb,
      shpP.W1, shpP.b1, shpP.lnw, shpP.lnb, shpP.W2, shpP.b2, f_cat);

  // motion features -> x_a fourier embedding (with cat_sum)
  sk_motion_feat<<<(MNT + 255) / 256, 256, 0, stream>>>(pos_a, head_a, f_mf);
  fourier(f_mf, MNT, 2, xaP, f_cat, MS, f_xa);

  // relative edge features -> fourier embeddings
  sk_edge_t_feat<<<(ET + 255) / 256, 256, 0, stream>>>(e_t, ET, pos_a, head_a, f_ef);
  fourier(f_ef, ET, 4, rtP, nullptr, 0, f_rt);
  sk_edge_p_feat<<<(EP + 255) / 256, 256, 0, stream>>>(e_p, EP, pos_pl, orient_pl, pos_a, head_a, f_ef);
  fourier(f_ef, EP, 3, rpP, nullptr, 0, f_rp);
  sk_edge_a_feat<<<(EA + 255) / 256, 256, 0, stream>>>(e_a, EA, pos_a, head_a, f_ef);
  fourier(f_ef, EA, 3, raP, nullptr, 0, f_ra);

  // fusion MLP: concat(token_emb, x_a) -> x
  sk_fuse_concat<<<(unsigned)(((size_t)MNT * 256 + 255) / 256), 256, 0, stream>>>(
      f_tab, atype, ati, f_xa, f_c2);
  gemm(f_c2, fusP.W1, fusP.b1, nullptr, f_q, MNT, MD, 256, 0, 0);
  ln(f_q, fusP.lnw, fusP.lnb, f_q, MNT, 1);
  gemm(f_q, fusP.W2, fusP.b2, nullptr, f_x, MNT, MD, MD, 0, 0);

  // stacked attention layers
  for (int i = 0; i < 3; ++i) {
    attn(f_x,  MNT,  f_x, f_rt, e_t, ET, ttA, i);
    attn(x_pl, MNPL, f_x, f_rp, e_p, EP, ptA, i);
    attn(f_x,  MNT,  f_x, f_ra, e_a, EA, aaA, i);
  }

  // head MLP -> logits [N,S,TOK]
  gemm(f_x, headP.W1, headP.b1, nullptr, f_q, MNT, MD, MD, 0, 0);
  ln(f_q, headP.lnw, headP.lnb, f_q, MNT, 1);
  gemm(f_q, headP.W2, headP.b2, nullptr, (float*)d_out, MNT, MTOK, MD, 0, 0);
}
